// VSN_MultiheadAttention_Block_72189810311326
// MI455X (gfx1250) — matrix operand order repaired and checked by reading, not compiled, not run on hardware
//
#include <hip/hip_runtime.h>
#include <hip/hip_bf16.h>

typedef __attribute__((ext_vector_type(16))) __bf16 v16bf;
typedef __attribute__((ext_vector_type(8)))  float  v8f;

#define GEMM_RELU  1
#define GEMM_ACCUM 2

__device__ __forceinline__ unsigned short f32_to_bf16(float f) {
  unsigned int u = __builtin_bit_cast(unsigned int, f);
  u += 0x7FFFu + ((u >> 16) & 1u);              // round-to-nearest-even
  return (unsigned short)(u >> 16);
}
__device__ __forceinline__ float bf16_to_f32(unsigned short h) {
  unsigned int u = ((unsigned int)h) << 16;
  return __builtin_bit_cast(float, u);
}

// LDS byte offset of a __shared__ object (LDS aperture: low 32 bits = offset)
__device__ __forceinline__ unsigned lds_addr_of(const void* p) {
  return (unsigned)(unsigned long long)p;
}

// CDNA5 async global->LDS copy, ASYNCcnt-tracked, no VGPR data staging.
// INST_OFFSET is added to BOTH the global and the LDS address, so one
// (lds, global) pair drives a 32-byte transfer with two instructions.
__device__ __forceinline__ void async_b128(unsigned lds_off, const void* g) {
  asm volatile("global_load_async_to_lds_b128 %0, %1, off"
               :: "v"(lds_off), "v"(g) : "memory");
}
__device__ __forceinline__ void async_b128_o16(unsigned lds_off, const void* g) {
  asm volatile("global_load_async_to_lds_b128 %0, %1, off offset:16"
               :: "v"(lds_off), "v"(g) : "memory");
}
__device__ __forceinline__ void wait_async0() {
  asm volatile("s_wait_asynccnt 0x0" ::: "memory");
}

// ---------------------------------------------------------------------------
// bf16 WMMA GEMM: C = act(alpha * A[M,K] * Bt[N,K]^T + bias)  (Bt is N-major)
// Block tile 128x128, K-step 32, 8 waves (4 in M x 2 in N), wave tile 32x64.
// Double-buffered LDS fed by GLOBAL_LOAD_ASYNC_TO_LDS_B128.
// Requires M%128==0, N%128==0, K%32==0, lda/ldb multiples of 8.
// ---------------------------------------------------------------------------
__global__ __launch_bounds__(256) void gemm_bf16_wmma(
    const unsigned short* __restrict__ A,
    const unsigned short* __restrict__ Bt,
    const float* __restrict__ bias,
    const float* __restrict__ rowscale, int rowscale_stride,
    float* __restrict__ Cf, unsigned short* __restrict__ Cb,
    int M, int N, int K, int lda, int ldb, int ldc,
    long long strideA, long long strideB, long long strideC,
    float alpha, int flags)
{
  __shared__ __align__(16) unsigned short As[2][128 * 40];
  __shared__ __align__(16) unsigned short Bs[2][128 * 40];

  const int tid  = threadIdx.x;
  const int lane = tid & 31;
  const int wave = tid >> 5;
  const int wm   = wave & 3;          // wave row  -> M offset wm*32
  const int wn   = wave >> 2;         // wave col  -> N offset wn*64
  const int g    = lane >> 4;         // lane half group
  const int lr   = lane & 15;

  const int m0 = blockIdx.y * 128;
  const int n0 = blockIdx.x * 128;
  const long long z = blockIdx.z;

  const unsigned short* Az = A  + z * strideA;
  const unsigned short* Bz = Bt + z * strideB;

  v8f acc[2][4] = {};

  // async copy mapping: each thread owns one 32B half-row of A and of B
  const int cr = tid >> 1;            // 0..127 tile row
  const int ch = (tid & 1) * 16;      // 0 or 16 (elements)
  const unsigned short* gA0 = Az + (long long)(m0 + cr) * lda + ch;
  const unsigned short* gB0 = Bz + (long long)(n0 + cr) * ldb + ch;
  const unsigned lA[2] = { lds_addr_of(&As[0][cr * 40 + ch]),
                           lds_addr_of(&As[1][cr * 40 + ch]) };
  const unsigned lB[2] = { lds_addr_of(&Bs[0][cr * 40 + ch]),
                           lds_addr_of(&Bs[1][cr * 40 + ch]) };

  const int nt = K / 32;

  // prologue: tile 0 -> buffer 0
  async_b128    (lA[0], gA0);
  async_b128_o16(lA[0], gA0);
  async_b128    (lB[0], gB0);
  async_b128_o16(lB[0], gB0);

  for (int t = 0; t < nt; ++t) {
    const int cur = t & 1;
    wait_async0();                    // my tile-t transfers have landed
    __syncthreads();                  // everyone's landed; other buffer free

    if (t + 1 < nt) {                 // stream tile t+1 while computing t
      const int k1 = (t + 1) * 32;
      async_b128    (lA[cur ^ 1], gA0 + k1);
      async_b128_o16(lA[cur ^ 1], gA0 + k1);
      async_b128    (lB[cur ^ 1], gB0 + k1);
      async_b128_o16(lB[cur ^ 1], gB0 + k1);
    }

    // A fragment (16x32): lane (g,lr) holds row lr, K = {8g..8g+7, 16+8g..23+8g}
    v16bf fa[2];
#pragma unroll
    for (int i = 0; i < 2; ++i) {
      const uint4* p = (const uint4*)(&As[cur][(wm * 32 + i * 16 + lr) * 40]);
      union { uint4 u[2]; v16bf v; } u;
      u.u[0] = p[g];
      u.u[1] = p[2 + g];
      fa[i] = u.v;
    }
    // B fragment (32x16): lane (g,lr) holds col lr, K = {8g..8g+7, 16+8g..23+8g}
    v16bf fb[4];
#pragma unroll
    for (int j = 0; j < 4; ++j) {
      const uint4* p = (const uint4*)(&Bs[cur][(wn * 64 + j * 16 + lr) * 40]);
      union { uint4 u[2]; v16bf v; } u;
      u.u[0] = p[g];
      u.u[1] = p[2 + g];
      fb[j] = u.v;
    }
#pragma unroll
    for (int i = 0; i < 2; ++i)
#pragma unroll
      for (int j = 0; j < 4; ++j)
        acc[i][j] = __builtin_amdgcn_wmma_f32_16x16x32_bf16(
            false, fa[i], false, fb[j], (short)0, acc[i][j], false, false);
  }

  // Epilogue: C VGPR r holds row g*8+r, col = lr (per 16x16 tile)
#pragma unroll
  for (int i = 0; i < 2; ++i) {
    const int rbase = m0 + wm * 32 + i * 16 + g * 8;
#pragma unroll
    for (int j = 0; j < 4; ++j) {
      const int col = n0 + wn * 64 + j * 16 + lr;
      const float bv = bias ? bias[col] : 0.0f;
#pragma unroll
      for (int r = 0; r < 8; ++r) {
        const int row = rbase + r;
        float v = fmaf(alpha, acc[i][j][r], bv);
        if (flags & GEMM_RELU) v = fmaxf(v, 0.0f);
        const long long cidx = z * strideC + (long long)row * ldc + col;
        if (flags & GEMM_ACCUM) {
          Cf[cidx] += rowscale[(long long)row * rowscale_stride] * v;
        } else {
          if (Cf) Cf[cidx] = v;
          if (Cb) Cb[cidx] = f32_to_bf16(v);
        }
      }
    }
  }
}

// ---------------------------------------------------------------------------
// helpers
// ---------------------------------------------------------------------------
__global__ void k_f32_to_bf16(const float* __restrict__ in,
                              unsigned short* __restrict__ out, long long n) {
  long long i = (long long)blockIdx.x * blockDim.x + threadIdx.x;
  if (i < n) out[i] = f32_to_bf16(in[i]);
}

__global__ void k_fill_zero_f32(float* __restrict__ p, long long n) {
  long long i = (long long)blockIdx.x * blockDim.x + threadIdx.x;
  if (i < n) p[i] = 0.0f;
}

// out[b, n*K + k] = bf16(in[b, k*N + n])
__global__ void k_transpose_to_bf16(const float* __restrict__ in,
                                    unsigned short* __restrict__ out,
                                    int K, int N,
                                    long long in_bstride, long long out_bstride) {
  long long idx = (long long)blockIdx.x * blockDim.x + threadIdx.x;
  long long total = (long long)K * N;
  if (idx >= total) return;
  int k = (int)(idx / N);
  int n = (int)(idx % N);
  const float* inz = in + (long long)blockIdx.z * in_bstride;
  unsigned short* outz = out + (long long)blockIdx.z * out_bstride;
  outz[(long long)n * K + k] = f32_to_bf16(inz[idx]);
}

// gate head: g = gh[M,512] @ Wg2[512,16] + bg2 ; w = softmax(g)
__global__ __launch_bounds__(256) void k_gate2_softmax(
    const unsigned short* __restrict__ gh, const float* __restrict__ Wg2,
    const float* __restrict__ bg2, float* __restrict__ w, int M) {
  int m = blockIdx.x * blockDim.x + threadIdx.x;
  if (m >= M) return;
  float acc[16];
#pragma unroll
  for (int n = 0; n < 16; ++n) acc[n] = bg2[n];
  const unsigned short* row = gh + (long long)m * 512;
  for (int k = 0; k < 512; ++k) {
    float gv = bf16_to_f32(row[k]);
    const float* wr = Wg2 + k * 16;
#pragma unroll
    for (int n = 0; n < 16; ++n) acc[n] = fmaf(gv, wr[n], acc[n]);
  }
  float mx = acc[0];
#pragma unroll
  for (int n = 1; n < 16; ++n) mx = fmaxf(mx, acc[n]);
  float s = 0.0f;
#pragma unroll
  for (int n = 0; n < 16; ++n) { acc[n] = expf(acc[n] - mx); s += acc[n]; }
  float inv = 1.0f / s;
#pragma unroll
  for (int n = 0; n < 16; ++n) w[(long long)m * 16 + n] = acc[n] * inv;
}

// row softmax over width 1024, in-place f32 + bf16 copy; one block per row
__global__ __launch_bounds__(256) void k_softmax_1024(
    float* __restrict__ probs, unsigned short* __restrict__ probs_b) {
  __shared__ float red[256];
  const long long base = (long long)blockIdx.x * 1024;
  const int tid = threadIdx.x;
  float v[4];
#pragma unroll
  for (int i = 0; i < 4; ++i) v[i] = probs[base + tid + i * 256];
  float mx = fmaxf(fmaxf(v[0], v[1]), fmaxf(v[2], v[3]));
  red[tid] = mx; __syncthreads();
  for (int s = 128; s > 0; s >>= 1) {
    if (tid < s) red[tid] = fmaxf(red[tid], red[tid + s]);
    __syncthreads();
  }
  mx = red[0]; __syncthreads();
  float sum = 0.0f;
#pragma unroll
  for (int i = 0; i < 4; ++i) { v[i] = expf(v[i] - mx); sum += v[i]; }
  red[tid] = sum; __syncthreads();
  for (int s = 128; s > 0; s >>= 1) {
    if (tid < s) red[tid] += red[tid + s];
    __syncthreads();
  }
  float inv = 1.0f / red[0];
#pragma unroll
  for (int i = 0; i < 4; ++i) {
    float p = v[i] * inv;
    probs[base + tid + i * 256] = p;
    probs_b[base + tid + i * 256] = f32_to_bf16(p);
  }
}

// ---------------------------------------------------------------------------
extern "C" void kernel_launch(void* const* d_in, const int* in_sizes, int n_in,
                              void* d_out, int out_size, void* d_ws, size_t ws_size,
                              hipStream_t stream) {
  (void)in_sizes; (void)n_in; (void)out_size; (void)ws_size;
  const float* x   = (const float*)d_in[0];
  const float* W1  = (const float*)d_in[1];
  const float* b1  = (const float*)d_in[2];
  const float* W2  = (const float*)d_in[3];
  const float* b2  = (const float*)d_in[4];
  const float* Wg1 = (const float*)d_in[5];
  const float* bg1 = (const float*)d_in[6];
  const float* Wg2 = (const float*)d_in[7];
  const float* bg2 = (const float*)d_in[8];
  const float* Wq  = (const float*)d_in[9];
  const float* bq  = (const float*)d_in[10];
  const float* Wk  = (const float*)d_in[11];
  const float* bk  = (const float*)d_in[12];
  const float* Wv  = (const float*)d_in[13];
  const float* bv  = (const float*)d_in[14];
  const float* Wo  = (const float*)d_in[15];
  const float* bo  = (const float*)d_in[16];

  constexpr int B = 8, S = 1024, F = 16, D = 256, H = 512;
  constexpr int M = B * S;      // 8192 tokens
  constexpr int FD = F * D;     // 4096

  float* out_attn  = (float*)d_out;                         // [M, D]
  float* out_w     = out_attn + (long long)M * D;           // [M, F]
  float* out_probs = out_w + (long long)M * F;              // [B, S, S]

  // workspace carve-up (256B aligned chunks)
  char* wsp = (char*)d_ws;
  auto alloc_us = [&](long long n) {
    unsigned short* p = (unsigned short*)wsp;
    wsp += ((n * 2 + 255) / 256) * 256; return p;
  };
  auto alloc_f = [&](long long n) {
    float* p = (float*)wsp;
    wsp += ((n * 4 + 255) / 256) * 256; return p;
  };

  unsigned short* xb    = alloc_us((long long)M * FD);   // x in bf16
  unsigned short* W1t   = alloc_us((long long)F * H * D);
  unsigned short* W2t   = alloc_us((long long)F * D * H);
  unsigned short* Wg1t  = alloc_us((long long)H * FD);
  unsigned short* Wqt   = alloc_us((long long)D * D);
  unsigned short* Wkt   = alloc_us((long long)D * D);
  unsigned short* Wvt   = alloc_us((long long)D * D);
  unsigned short* Wot   = alloc_us((long long)D * D);
  unsigned short* h_tmp = alloc_us((long long)M * H);    // per-feature hidden
  unsigned short* g_h   = alloc_us((long long)M * H);    // gate hidden
  float*          vsn_f = alloc_f((long long)M * D);
  unsigned short* vsn_b = alloc_us((long long)M * D);
  unsigned short* q_b   = alloc_us((long long)M * D);
  unsigned short* k_b   = alloc_us((long long)M * D);
  float*          v_f   = alloc_f((long long)M * D);
  unsigned short* vt_b  = alloc_us((long long)M * D);    // [B, D, S]
  unsigned short* pr_b  = alloc_us((long long)B * S * S);
  unsigned short* ctx_b = alloc_us((long long)M * D);

  auto gemm = [&](const unsigned short* A_, const unsigned short* Bt_,
                  const float* bias_, const float* rs_, int rs_stride,
                  float* Cf_, unsigned short* Cb_,
                  int Mm, int Nn, int Kk, int lda, int ldb, int ldc,
                  long long sA, long long sB, long long sC, int batch,
                  float alpha, int flags) {
    dim3 grid((unsigned)(Nn / 128), (unsigned)(Mm / 128), (unsigned)batch);
    gemm_bf16_wmma<<<grid, 256, 0, stream>>>(
        A_, Bt_, bias_, rs_, rs_stride, Cf_, Cb_,
        Mm, Nn, Kk, lda, ldb, ldc, sA, sB, sC, alpha, flags);
  };

  // --- 1. precision conversion + weight transposition -------------------
  {
    long long n = (long long)M * FD;
    k_f32_to_bf16<<<(unsigned)((n + 255) / 256), 256, 0, stream>>>(x, xb, n);
  }
  k_transpose_to_bf16<<<dim3((D * H + 255) / 256, 1, F), 256, 0, stream>>>(
      W1, W1t, D, H, (long long)D * H, (long long)H * D);
  k_transpose_to_bf16<<<dim3((H * D + 255) / 256, 1, F), 256, 0, stream>>>(
      W2, W2t, H, D, (long long)H * D, (long long)D * H);
  k_transpose_to_bf16<<<dim3((FD * H + 255) / 256, 1, 1), 256, 0, stream>>>(
      Wg1, Wg1t, FD, H, 0, 0);
  k_transpose_to_bf16<<<dim3((D * D + 255) / 256, 1, 1), 256, 0, stream>>>(
      Wq, Wqt, D, D, 0, 0);
  k_transpose_to_bf16<<<dim3((D * D + 255) / 256, 1, 1), 256, 0, stream>>>(
      Wk, Wkt, D, D, 0, 0);
  k_transpose_to_bf16<<<dim3((D * D + 255) / 256, 1, 1), 256, 0, stream>>>(
      Wv, Wvt, D, D, 0, 0);
  k_transpose_to_bf16<<<dim3((D * D + 255) / 256, 1, 1), 256, 0, stream>>>(
      Wo, Wot, D, D, 0, 0);

  // --- 2. gate: relu(flat @ Wg1 + bg1) @ Wg2 + bg2 -> softmax -> w ------
  gemm(xb, Wg1t, bg1, nullptr, 0, nullptr, g_h,
       M, H, FD, FD, FD, H, 0, 0, 0, 1, 1.0f, GEMM_RELU);
  k_gate2_softmax<<<M / 256, 256, 0, stream>>>(g_h, Wg2, bg2, out_w, M);

  // --- 3. per-feature expert MLPs fused with weighted sum ---------------
  k_fill_zero_f32<<<(unsigned)(((long long)M * D + 255) / 256), 256, 0, stream>>>(
      vsn_f, (long long)M * D);
  for (int f = 0; f < F; ++f) {
    // h_f = relu(x[:,:,f,:] @ W1_f + b1_f)
    gemm(xb + f * D, W1t + (long long)f * H * D, b1 + (long long)f * H,
         nullptr, 0, nullptr, h_tmp,
         M, H, D, FD, D, H, 0, 0, 0, 1, 1.0f, GEMM_RELU);
    // vsn += w[:,f] * (h_f @ W2_f + b2_f)
    gemm(h_tmp, W2t + (long long)f * D * H, b2 + (long long)f * D,
         out_w + f, F, vsn_f, nullptr,
         M, D, H, H, H, D, 0, 0, 0, 1, 1.0f, GEMM_ACCUM);
  }
  k_f32_to_bf16<<<(unsigned)(((long long)M * D + 255) / 256), 256, 0, stream>>>(
      vsn_f, vsn_b, (long long)M * D);

  // --- 4. q/k/v projections ---------------------------------------------
  gemm(vsn_b, Wqt, bq, nullptr, 0, nullptr, q_b,
       M, D, D, D, D, D, 0, 0, 0, 1, 1.0f, 0);
  gemm(vsn_b, Wkt, bk, nullptr, 0, nullptr, k_b,
       M, D, D, D, D, D, 0, 0, 0, 1, 1.0f, 0);
  gemm(vsn_b, Wvt, bv, nullptr, 0, v_f, nullptr,
       M, D, D, D, D, D, 0, 0, 0, 1, 1.0f, 0);
  // vt[b, d, s] = v[b, s, d]
  k_transpose_to_bf16<<<dim3((S * D + 255) / 256, 1, B), 256, 0, stream>>>(
      v_f, vt_b, S, D, (long long)S * D, (long long)S * D);

  // --- 5. scores = q @ k^T / sqrt(256), batched over B ------------------
  gemm(q_b, k_b, nullptr, nullptr, 0, out_probs, nullptr,
       S, S, D, D, D, S,
       (long long)S * D, (long long)S * D, (long long)S * S, B, 0.0625f, 0);

  // --- 6. softmax over keys (in-place, plus bf16 copy) ------------------
  k_softmax_1024<<<B * S, 256, 0, stream>>>(out_probs, pr_b);

  // --- 7. ctx = probs @ v, batched over B -------------------------------
  gemm(pr_b, vt_b, nullptr, nullptr, 0, nullptr, ctx_b,
       S, D, S, S, S, D,
       (long long)S * S, (long long)D * S, (long long)S * D, B, 1.0f, 0);

  // --- 8. out = ctx @ Wo + bo -------------------------------------------
  gemm(ctx_b, Wot, bo, nullptr, 0, out_attn, nullptr,
       M, D, D, D, D, D, 0, 0, 0, 1, 1.0f, 0);
}
